// Dense_test_1layer_40510131536155
// MI455X (gfx1250) — matrix operand order repaired and checked by reading, not compiled, not run on hardware
//
#include <hip/hip_runtime.h>

// ---- problem constants (from reference) ----
#define B_SZ   2048
#define T_SZ   1000
#define I_SZ   40      // inputs
#define H_SZ   16      // hidden neurons
#define BR     2       // branches
#define NB     32      // H*BR
#define KP     64      // K padded to 2 x 32 WMMA chunks
#define O_SZ   2

typedef __attribute__((ext_vector_type(16))) _Float16 v16h;
typedef __attribute__((ext_vector_type(8)))  float    v8f;

// ---------------------------------------------------------------------------
// Init: build masked f16 weights Wm[32][64] (K zero-padded), alpha/beta decays,
// zero the atomic reduction slots, and write the constant `total` output.
// Runs every call => graph-replay deterministic.
// ---------------------------------------------------------------------------
__global__ void snn_init_kernel(const float* __restrict__ W1,
                                const float* __restrict__ tau_m,
                                const float* __restrict__ tau_n,
                                const float* __restrict__ mask,
                                float* __restrict__ d_out,
                                _Float16* __restrict__ wm,
                                float* __restrict__ alpha,
                                float* __restrict__ beta)
{
  int tid = threadIdx.x;
  for (int i = tid; i < NB * KP; i += blockDim.x) {
    int n = i / KP, k = i % KP;
    float v = (k < I_SZ) ? W1[n * I_SZ + k] * mask[n * I_SZ + k] : 0.0f;
    wm[i] = (_Float16)v;
  }
  if (tid < H_SZ) alpha[tid] = 1.0f / (1.0f + __expf(-tau_m[tid]));
  if (tid < NB)   beta[tid]  = 1.0f / (1.0f + __expf(-tau_n[tid]));
  if (tid == 0) {
    d_out[0] = 0.0f;                                   // loss
    d_out[1 + (size_t)B_SZ * T_SZ * O_SZ] = 0.0f;      // correct
    int cnt = 0;
    for (int t = 0; t < T_SZ; ++t)
      cnt += ((t > 10) && (((t - 10) % 15) > 5)) ? 1 : 0;
    d_out[2 + (size_t)B_SZ * T_SZ * O_SZ] = (float)cnt * (float)B_SZ; // total
  }
}

// ---------------------------------------------------------------------------
// Main: one wave per 16-batch-row tile; T recurrence in registers.
//   - x fragment loaded straight from global in WMMA A-layout (6x float4/lane,
//     software-pipelined one step ahead, prefetch two ahead)
//   - cur[16x32] = x_t @ Wm^T via 4x v_wmma_f32_16x16x32_f16
//   - dendrite decay in WMMA D-layout (per-lane scalar beta)
//   - 2KB LDS transpose; tail split over all 32 lanes: lane owns
//     (row = lane%16, h-range = 8*(lane/16)..+8); partial readouts combined
//     with two xor-16 shuffles; CE tail on lanes 0..15
// ---------------------------------------------------------------------------
__global__ __launch_bounds__(32)
void snn_step_kernel(const float* __restrict__ x,
                     const int*   __restrict__ target,
                     const _Float16* __restrict__ wm,
                     const float* __restrict__ alpha,
                     const float* __restrict__ beta,
                     const float* __restrict__ W2,
                     const float* __restrict__ b2,
                     float* __restrict__ d_out)
{
  __shared__ float tb[16][NB];             // [M][nb] transpose buffer, 2 KB

  const int lane  = threadIdx.x;           // 0..31 (wave32)
  const int m     = lane & 15;
  const int half  = lane >> 4;
  const int bBase = blockIdx.x * 16;

  // ---- B-matrix (Wm) resident in registers for the whole T loop ----
  // B layout (f16 32x16): N = lane%16, K = comp + 16*half (+32 for chunk 1)
  const int klane = half * 16;
  const _Float16* r0 = wm + (size_t)m * KP;          // tile0: nb = m
  const _Float16* r1 = wm + (size_t)(m + 16) * KP;   // tile1: nb = m+16
  const v16h b00 = *(const v16h*)(r0 + klane);
  const v16h b01 = *(const v16h*)(r0 + 32 + klane);
  const v16h b10 = *(const v16h*)(r1 + klane);
  const v16h b11 = *(const v16h*)(r1 + 32 + klane);

  // dendrite decay constants in D layout (N = lane%16)
  const float betaA  = beta[m];
  const float betaB  = beta[m + 16];
  const float ombA   = 1.0f - betaA;
  const float ombB   = 1.0f - betaB;

  // tail constants: this lane handles h = 8*half + j, j = 0..7
  float alphaJ[8], omaJ[8], w2aJ[8], w2bJ[8];
#pragma unroll
  for (int j = 0; j < 8; ++j) {
    const int h = 8 * half + j;
    alphaJ[j] = alpha[h];
    omaJ[j]   = 1.0f - alphaJ[j];
    w2aJ[j]   = W2[h];
    w2bJ[j]   = W2[H_SZ + h];
  }
  const float bb0 = b2[0], bb1 = b2[1];

  // recurrent state
  v8f dA = {}, dB = {};                    // dendrite, D layout, nb tiles 0/1
  float memJ[8], spkJ[8];                  // membrane/spike, 8 h per lane
#pragma unroll
  for (int j = 0; j < 8; ++j) { memJ[j] = 0.0f; spkJ[j] = 0.0f; }
  float accLoss = 0.0f, accCorr = 0.0f;

  // A-fragment addressing: lane reads row m of x_t, K-base kb = 16*half
  const int   kb   = half * 16;
  const float hmsk = (half == 0) ? 1.0f : 0.0f;   // mask for K=32..39 chunk
  const float* rowp = x + (size_t)(bBase + m) * T_SZ * I_SZ;
  const int*   tptr = target + (size_t)(bBase + m) * T_SZ;
  float* outp = d_out + 1 + (size_t)(bBase + m) * T_SZ * O_SZ;

  // LDS write/read bases
  float* tw = &tb[8 * half][m];            // stores: tw[c*NB] / tw[c*NB+16]
  const float* tr = &tb[m][klane];         // reads: own row, own nb half

  // preload t = 0 fragment (6x float4 per lane)
  float4 c0 = *(const float4*)(rowp + kb);
  float4 c1 = *(const float4*)(rowp + kb + 4);
  float4 c2 = *(const float4*)(rowp + kb + 8);
  float4 c3 = *(const float4*)(rowp + kb + 12);
  float4 c4 = *(const float4*)(rowp + 32);
  float4 c5 = *(const float4*)(rowp + 36);

  int ph = 0;                              // (t-10) % 15, maintained by wrap
  for (int t = 0; t < T_SZ; ++t) {
    // ---- software pipeline: issue t+1 loads, prefetch t+2 ----
    float4 n0 = c0, n1 = c1, n2 = c2, n3 = c3, n4 = c4, n5 = c5;
    if (t + 1 < T_SZ) {
      const float* nrp = rowp + I_SZ;
      n0 = *(const float4*)(nrp + kb);
      n1 = *(const float4*)(nrp + kb + 4);
      n2 = *(const float4*)(nrp + kb + 8);
      n3 = *(const float4*)(nrp + kb + 12);
      n4 = *(const float4*)(nrp + 32);
      n5 = *(const float4*)(nrp + 36);
      if (t + 2 < T_SZ)                              // both 128B lines of row
        __builtin_prefetch(rowp + 2 * I_SZ + half * 32, 0, 3);
    }

    // ---- build A fragments (f16 16x32 layout) from current registers ----
    union { v16h v; _Float16 h[16]; } A0, A1;
    A0.h[0]  = (_Float16)c0.x; A0.h[1]  = (_Float16)c0.y;
    A0.h[2]  = (_Float16)c0.z; A0.h[3]  = (_Float16)c0.w;
    A0.h[4]  = (_Float16)c1.x; A0.h[5]  = (_Float16)c1.y;
    A0.h[6]  = (_Float16)c1.z; A0.h[7]  = (_Float16)c1.w;
    A0.h[8]  = (_Float16)c2.x; A0.h[9]  = (_Float16)c2.y;
    A0.h[10] = (_Float16)c2.z; A0.h[11] = (_Float16)c2.w;
    A0.h[12] = (_Float16)c3.x; A0.h[13] = (_Float16)c3.y;
    A0.h[14] = (_Float16)c3.z; A0.h[15] = (_Float16)c3.w;
    A1.h[0]  = (_Float16)(c4.x * hmsk); A1.h[1] = (_Float16)(c4.y * hmsk);
    A1.h[2]  = (_Float16)(c4.z * hmsk); A1.h[3] = (_Float16)(c4.w * hmsk);
    A1.h[4]  = (_Float16)(c5.x * hmsk); A1.h[5] = (_Float16)(c5.y * hmsk);
    A1.h[6]  = (_Float16)(c5.z * hmsk); A1.h[7] = (_Float16)(c5.w * hmsk);
#pragma unroll
    for (int i = 8; i < 16; ++i) A1.h[i] = (_Float16)0.0f;   // K >= 40 pad

    // ---- cur = x_t @ Wm^T : 2 N-tiles x 2 K-chunks ----
    v8f cA = {};
    cA = __builtin_amdgcn_wmma_f32_16x16x32_f16(false, A0.v, false, b00, (short)0, cA, false, false);
    cA = __builtin_amdgcn_wmma_f32_16x16x32_f16(false, A1.v, false, b01, (short)0, cA, false, false);
    v8f cB = {};
    cB = __builtin_amdgcn_wmma_f32_16x16x32_f16(false, A0.v, false, b10, (short)0, cB, false, false);
    cB = __builtin_amdgcn_wmma_f32_16x16x32_f16(false, A1.v, false, b11, (short)0, cB, false, false);

    // ---- dendritic low-pass in D layout: d = beta*d + (1-beta)*cur ----
    dA = betaA * dA + ombA * cA;
    dB = betaB * dB + ombB * cB;

    // ---- transpose to [M][nb] via LDS ----
    __syncthreads();
#pragma unroll
    for (int c = 0; c < 8; ++c) {
      tw[c * NB]      = dA[c];
      tw[c * NB + 16] = dB[c];
    }
    __syncthreads();

    // ---- split tail: lane handles row m, h = 8*half + j ----
    float l0p = 0.0f, l1p = 0.0f;
#pragma unroll
    for (int q = 0; q < 4; ++q) {
      const float4 v = *(const float4*)(tr + q * 4);   // nb = klane+4q..+3
      const int j0 = 2 * q, j1 = 2 * q + 1;
      float lv0 = v.x + v.y;                           // branch sum
      float lv1 = v.z + v.w;
      float m0 = (memJ[j0] - spkJ[j0]) * alphaJ[j0] + omaJ[j0] * lv0;
      float m1 = (memJ[j1] - spkJ[j1]) * alphaJ[j1] + omaJ[j1] * lv1;
      float s0 = (m0 > 1.0f) ? 1.0f : 0.0f;            // VTH = 1
      float s1 = (m1 > 1.0f) ? 1.0f : 0.0f;
      memJ[j0] = m0; memJ[j1] = m1;
      spkJ[j0] = s0; spkJ[j1] = s1;
      l0p = fmaf(s0, w2aJ[j0], fmaf(s1, w2aJ[j1], l0p));
      l1p = fmaf(s0, w2bJ[j0], fmaf(s1, w2bJ[j1], l1p));
    }
    // combine the two h-halves (lanes L and L^16 share a batch row)
    const float l0 = bb0 + l0p + __shfl_xor(l0p, 16, 32);
    const float l1 = bb1 + l1p + __shfl_xor(l1p, 16, 32);

    const bool flag = (t > 10) && (ph > 5);
    if (lane < 16) {
      // logits out (d_out+1 is only 4B aligned -> two b32 stores)
      outp[(size_t)t * 2]     = l0;
      outp[(size_t)t * 2 + 1] = l1;
      if (flag) {
        // double softmax CE (reference log_softmax(softmax(logits)))
        float mx = fmaxf(l0, l1);
        float e0 = __expf(l0 - mx), e1 = __expf(l1 - mx);
        float inv = 1.0f / (e0 + e1);
        float p0 = e0 * inv, p1 = e1 * inv;
        float mp  = fmaxf(p0, p1);
        float lse = mp + __logf(__expf(p0 - mp) + __expf(p1 - mp));
        int   tg  = tptr[t];
        accLoss += lse - (tg ? p1 : p0);
        accCorr += ((p1 > p0 ? 1 : 0) == tg) ? 1.0f : 0.0f;
      }
    }

    // ---- rotate pipeline / phase ----
    if (t >= 10) ph = (ph == 14) ? 0 : ph + 1;
    rowp += I_SZ;
    c0 = n0; c1 = n1; c2 = n2; c3 = n3; c4 = n4; c5 = n5;
  }

  if (lane < 16) {
    atomicAdd(&d_out[0], accLoss * (1.0f / (float)B_SZ));
    atomicAdd(&d_out[1 + (size_t)B_SZ * T_SZ * O_SZ], accCorr);
  }
}

// ---------------------------------------------------------------------------
extern "C" void kernel_launch(void* const* d_in, const int* in_sizes, int n_in,
                              void* d_out, int out_size, void* d_ws, size_t ws_size,
                              hipStream_t stream)
{
  const float* x      = (const float*)d_in[0];  // [B,T,I]
  const int*   target = (const int*)  d_in[1];  // [B,T]
  const float* W1     = (const float*)d_in[2];  // [NB,I]
  const float* tau_m  = (const float*)d_in[3];  // [H]
  const float* tau_n  = (const float*)d_in[4];  // [H,BR]
  const float* mask   = (const float*)d_in[5];  // [NB,I]
  const float* W2     = (const float*)d_in[6];  // [O,H]
  const float* b2     = (const float*)d_in[7];  // [O]
  float* out = (float*)d_out;

  // workspace: Wm f16 [32][64] (4096 B) | alpha[16] | beta[32]
  _Float16* wm    = (_Float16*)d_ws;
  float*    alpha = (float*)((char*)d_ws + NB * KP * sizeof(_Float16));
  float*    beta  = alpha + H_SZ;

  snn_init_kernel<<<1, 256, 0, stream>>>(W1, tau_m, tau_n, mask, out, wm, alpha, beta);
  snn_step_kernel<<<B_SZ / 16, 32, 0, stream>>>(x, target, wm, alpha, beta, W2, b2, out);
}
